// coRNN2_27857157882575
// MI455X (gfx1250) — matrix operand order repaired and checked by reading, not compiled, not run on hardware
//
#include <hip/hip_runtime.h>
#include <cstdint>
#include <cstddef>

typedef __attribute__((ext_vector_type(16))) _Float16 v16h;
typedef __attribute__((ext_vector_type(8)))  _Float16 v8h;
typedef __attribute__((ext_vector_type(8)))  float    v8f;

constexpr int cN_INP = 3, cN_HID = 64, cN_OUT = 10;
constexpr int cSEQ = 8, cROLL = 16;
constexpr int cH = 12, cW = 10, cPIX = cH * cW;     // 120
constexpr int cB = 16;
constexpr int cKDIM = 2 * cN_HID * 9;               // 1152
constexpr float cDT = 0.042f, cGAMMA = 2.7f, cEPS = 4.7f;

// ---------------------------------------------------------------------------
// Weight re-layout: pw_w[o][c][ki][kj][h][w] f32  ->  w_h[p][o][k] f16
// with p = h*10+w, k = c*9 + ki*3 + kj  (matches einsum reduction order)
// ---------------------------------------------------------------------------
__global__ void coRNN_wconv(const float* __restrict__ pw_w,
                            _Float16* __restrict__ w_h) {
  int i = blockIdx.x * blockDim.x + threadIdx.x;
  const int TOT = cPIX * cN_HID * cKDIM;
  if (i >= TOT) return;
  int k = i % cKDIM;
  int o = (i / cKDIM) % cN_HID;
  int p = i / (cKDIM * cN_HID);
  int c = k / 9, ij = k % 9;
  int ki = ij / 3, kj = ij % 3;
  size_t src = ((((size_t)o * (2 * cN_HID) + c) * 3 + ki) * 3 + kj) * cPIX + p;
  w_h[i] = (_Float16)pw_w[src];
}

// ---------------------------------------------------------------------------
// Readin conv (edge padding), writes f32 hy/hz state + f16 cat=[hz,hy] operand
// state layout: [channel][h][w][b] (b fastest);  cat: [c' (0..127)][h][w][b]
// ---------------------------------------------------------------------------
__global__ void coRNN_readin(const float* __restrict__ x,
                             const float* __restrict__ rw,
                             const float* __restrict__ rb,
                             float* __restrict__ hy, float* __restrict__ hz,
                             _Float16* __restrict__ cat) {
  int i = blockIdx.x * blockDim.x + threadIdx.x;
  const int TOT = 2 * cN_HID * cPIX * cB;           // 245760
  if (i >= TOT) return;
  int b = i % cB;
  int p = (i / cB) % cPIX;
  int c = i / (cB * cPIX);
  int h = p / cW, w = p % cW;
  float acc = rb[c];
  for (int t = 0; t < cSEQ; ++t)
    for (int ii = 0; ii < cN_INP; ++ii) {
      int ci = t * cN_INP + ii;
      for (int kh = 0; kh < 3; ++kh)
        for (int kw = 0; kw < 3; ++kw) {
          int hh = h + kh - 1; hh = hh < 0 ? 0 : (hh > cH - 1 ? cH - 1 : hh);
          int ww = w + kw - 1; ww = ww < 0 ? 0 : (ww > cW - 1 ? cW - 1 : ww);
          float xv = x[(((size_t)t * cB + b) * cN_INP + ii) * cPIX + hh * cW + ww];
          float wv = rw[(((size_t)c * (cSEQ * cN_INP) + ci) * 3 + kh) * 3 + kw];
          acc += xv * wv;
        }
    }
  if (c < cN_HID) {        // hy channel c -> cat channel 64+c
    hy[((size_t)c * cPIX + p) * cB + b] = acc;
    cat[(((size_t)(cN_HID + c)) * cPIX + p) * cB + b] = (_Float16)acc;
  } else {                 // hz channel c-64 -> cat channel c-64
    int c0 = c - cN_HID;
    hz[((size_t)c0 * cPIX + p) * cB + b] = acc;
    cat[((size_t)c0 * cPIX + p) * cB + b] = (_Float16)acc;
  }
}

// ---------------------------------------------------------------------------
// One rollout step. One workgroup per pixel (120 WGs, 4 waves).
// Per pixel: GEMM  pre[b,o] = A[b,k] * W[o,k]^T   (M=16, N=64, K=1152)
// via V_WMMA_F32_16X16X32_F16, one 16x16 N-tile per wave, then in-register
// coRNN Euler update; f32 state + f16 cat double-buffered across launches.
// ---------------------------------------------------------------------------
__global__ __launch_bounds__(128)
void coRNN_step(const _Float16* __restrict__ cat_in,
                const float* __restrict__ hy_in, const float* __restrict__ hz_in,
                const _Float16* __restrict__ w_h, const float* __restrict__ pw_b,
                float* __restrict__ hy_out, float* __restrict__ hz_out,
                _Float16* __restrict__ cat_out) {
  __shared__ __align__(16) _Float16 A[cB * cKDIM];   // [b][k], 36864 bytes
  const int p = blockIdx.x;
  const int h = p / cW, w = p % cW;
  const int tid = threadIdx.x;

  // Build A tile: A[b][c*9+ij] = cat_in[c][h+di][w+dj][b], zero-padded.
  for (int e = tid; e < cKDIM; e += 128) {           // 9 uniform iterations
    int c = e / 9, ij = e % 9;
    int hh = h + ij / 3 - 1, ww = w + ij % 3 - 1;
    v8h lo = {}, hi = {};
    if (hh >= 0 && hh < cH && ww >= 0 && ww < cW) {
      const v8h* s = (const v8h*)(cat_in + ((size_t)c * cPIX + hh * cW + ww) * cB);
      lo = s[0]; hi = s[1];
    }
#pragma unroll
    for (int r = 0; r < 8; ++r) {
      A[r * cKDIM + e]       = lo[r];
      A[(r + 8) * cKDIM + e] = hi[r];
    }
  }
  __syncthreads();

  const int lane = tid & 31;
  const int wv   = tid >> 5;          // 0..3  -> out-channel tile
  const int n    = lane & 15;
  const int up   = lane >> 4;         // 0/1 half-wave
  const int o    = wv * 16 + n;       // output channel this lane owns

  const _Float16* wrow = w_h + ((size_t)p * cN_HID + o) * cKDIM;
  v8f acc = {};
#pragma unroll 4
  for (int kb = 0; kb < cKDIM / 32; ++kb) {
    // A fragment: row m = n, K halves {0..7,16..23} + up*8  (ISA 16b-A layout)
    const _Float16* ap = A + (size_t)n * cKDIM + kb * 32 + up * 8;
    v16h a;
    ((uint4*)&a)[0] = ((const uint4*)ap)[0];
    ((uint4*)&a)[1] = *(const uint4*)(ap + 16);
    // B fragment: col o, K halves {0..7,16..23} + up*8, matching A
    const _Float16* bp = wrow + kb * 32 + up * 8;
    v16h bm;
    ((uint4*)&bm)[0] = ((const uint4*)bp)[0];
    ((uint4*)&bm)[1] = *(const uint4*)(bp + 16);
    __builtin_prefetch(bp + 256, 0, 1);             // global_prefetch_b8
    acc = __builtin_amdgcn_wmma_f32_16x16x32_f16(false, a, false, bm,
                                                 (short)0, acc, false, false);
  }

  // C tile rows: b = r + up*8 ; state layout [o][p][b] -> contiguous in r
  const float  bias = pw_b[(size_t)o * cPIX + p];
  const size_t sb   = ((size_t)o * cPIX + p) * cB + up * 8;
  v8h yh, zh;
#pragma unroll
  for (int r = 0; r < 8; ++r) {
    float hyv = hy_in[sb + r];
    float hzv = hz_in[sb + r];
    float pre = acc[r] + bias;
    float zn  = hzv + cDT * (tanhf(pre) - cGAMMA * hyv - cEPS * hzv);
    float yn  = hyv + cDT * zn;
    hy_out[sb + r] = yn;
    hz_out[sb + r] = zn;
    zh[r] = (_Float16)zn;
    yh[r] = (_Float16)yn;
  }
  *(v8h*)(cat_out + ((size_t)o            * cPIX + p) * cB + up * 8) = zh; // cat[0:64]=hz
  *(v8h*)(cat_out + ((size_t)(cN_HID + o) * cPIX + p) * cB + up * 8) = yh; // cat[64:]=hy
}

// ---------------------------------------------------------------------------
// Readout: out[b,o,h,w] = sum_c hy[c,h,w,b]*W[o,c] + b[o]
// ---------------------------------------------------------------------------
__global__ void coRNN_readout(const float* __restrict__ hy,
                              const float* __restrict__ rw,
                              const float* __restrict__ rb,
                              float* __restrict__ out) {
  int i = blockIdx.x * blockDim.x + threadIdx.x;
  if (i >= cB * cN_OUT * cPIX) return;
  int p = i % cPIX;
  int o = (i / cPIX) % cN_OUT;
  int b = i / (cPIX * cN_OUT);
  float acc = rb[o];
  for (int c = 0; c < cN_HID; ++c)
    acc += hy[((size_t)c * cPIX + p) * cB + b] * rw[o * cN_HID + c];
  out[i] = acc;      // i == ((b*10+o)*120 + p), matches (B,10,12,10) layout
}

// ---------------------------------------------------------------------------
extern "C" void kernel_launch(void* const* d_in, const int* in_sizes, int n_in,
                              void* d_out, int out_size, void* d_ws, size_t ws_size,
                              hipStream_t stream) {
  (void)in_sizes; (void)n_in; (void)out_size; (void)ws_size;
  const float* x   = (const float*)d_in[0];
  const float* riw = (const float*)d_in[1];
  const float* rib = (const float*)d_in[2];
  const float* pww = (const float*)d_in[3];
  const float* pwb = (const float*)d_in[4];
  const float* row = (const float*)d_in[5];
  const float* rob = (const float*)d_in[6];
  float* out = (float*)d_out;

  char* ws = (char*)d_ws;
  const size_t WH_BYTES  = (size_t)cPIX * cN_HID * cKDIM * sizeof(_Float16); // ~16.9 MB
  const size_t ST_BYTES  = (size_t)cN_HID * cPIX * cB * sizeof(float);       // 480 KB
  const size_t CAT_BYTES = (size_t)2 * cN_HID * cPIX * cB * sizeof(_Float16);// 480 KB
  _Float16* w_h  = (_Float16*)ws;
  float*    hyA  = (float*)(ws + WH_BYTES);
  float*    hzA  = (float*)(ws + WH_BYTES + 1 * ST_BYTES);
  float*    hyB  = (float*)(ws + WH_BYTES + 2 * ST_BYTES);
  float*    hzB  = (float*)(ws + WH_BYTES + 3 * ST_BYTES);
  _Float16* catA = (_Float16*)(ws + WH_BYTES + 4 * ST_BYTES);
  _Float16* catB = (_Float16*)(ws + WH_BYTES + 4 * ST_BYTES + CAT_BYTES);

  {
    int tot = cPIX * cN_HID * cKDIM;
    coRNN_wconv<<<(tot + 255) / 256, 256, 0, stream>>>(pww, w_h);
  }
  {
    int tot = 2 * cN_HID * cPIX * cB;
    coRNN_readin<<<(tot + 255) / 256, 256, 0, stream>>>(x, riw, rib, hyA, hzA, catA);
  }
  for (int s = 0; s < cROLL; ++s) {
    bool even = (s & 1) == 0;
    coRNN_step<<<cPIX, 128, 0, stream>>>(
        even ? catA : catB, even ? hyA : hyB, even ? hzA : hzB,
        w_h, pwb,
        even ? hyB : hyA, even ? hzB : hzA, even ? catB : catA);
  }
  {
    int tot = cB * cN_OUT * cPIX;  // final state lands in buffer A after 16 steps
    coRNN_readout<<<(tot + 255) / 256, 256, 0, stream>>>(hyA, row, rob, out);
  }
}
